// AttentionBlock_29506425323892
// MI455X (gfx1250) — matrix operand order repaired and checked by reading, not compiled, not run on hardware
//
#include <hip/hip_runtime.h>

// MI455X / gfx1250: wave32, WMMA 16x16x32 bf16 -> f32 accumulate,
// async global->LDS staging (ASYNCcnt) for the flash-attention K/V stream.
typedef __attribute__((ext_vector_type(16))) __bf16 v16bf;
typedef __attribute__((ext_vector_type(8)))  float  v8f;

#define C_DIM   256
#define HW_DIM  4096
#define B_DIM   8
#define NGROUPS 32
#define CPG     8
#define EPS_GN  1e-5f

__device__ __forceinline__ v8f zero8() {
  v8f z;
#pragma unroll
  for (int i = 0; i < 8; ++i) z[i] = 0.0f;
  return z;
}

// A fragment: 16x32 bf16, row-major A with leading dim lda.
// lane<16: M=lane, kb=0 ; lane>=16: M=lane-16, kb=8 (CDNA5 ISA 7.12.2)
__device__ __forceinline__ v16bf load_a_frag(const __bf16* __restrict__ A,
                                             size_t row0, int lda, int k0, int lane) {
  int m  = lane & 15;
  int kb = (lane < 16) ? 0 : 8;
  const __bf16* p = A + (row0 + (size_t)m) * (size_t)lda + (size_t)(k0 + kb);
  v16bf a;
#pragma unroll
  for (int t = 0; t < 8; ++t) { a[t] = p[t]; a[t + 8] = p[16 + t]; }
  return a;
}

// B fragment: 32x16 bf16, B given column-major as Bc[N][K] row-major (ldb = K stride).
// lane<16: N=lane, kb=0 ; lane>=16: N=lane-16, kb=8 (same K order as A frag)
__device__ __forceinline__ v16bf load_b_frag(const __bf16* __restrict__ Bc,
                                             size_t n0, int ldb, int k0, int lane) {
  int n  = lane & 15;
  int kb = (lane < 16) ? 0 : 8;
  const __bf16* p = Bc + (n0 + (size_t)n) * (size_t)ldb + (size_t)(k0 + kb);
  v16bf b;
#pragma unroll
  for (int t = 0; t < 8; ++t) { b[t] = p[t]; b[t + 8] = p[16 + t]; }
  return b;
}

__device__ __forceinline__ v8f wmma_bf16(v16bf a, v16bf b, v8f c) {
  return __builtin_amdgcn_wmma_f32_16x16x32_bf16(false, a, false, b, (short)0, c,
                                                 false, false);
}

// gfx1250 async copy global -> LDS, 16B per lane, tracked by ASYNCcnt.
__device__ __forceinline__ void async_load_b128(unsigned lds_off, const void* gptr) {
  asm volatile("global_load_async_to_lds_b128 %0, %1, off"
               :: "v"(lds_off), "v"(gptr) : "memory");
}
__device__ __forceinline__ void wait_async0() {
  asm volatile("s_wait_asynccnt 0x0" ::: "memory");
}

// ---------------- GroupNorm (fp32 stats) + bf16 transposed casts ----------------
__global__ void groupnorm_cast_kernel(const float* __restrict__ x,
                                      const float* __restrict__ gamma,
                                      const float* __restrict__ beta,
                                      __bf16* __restrict__ hnT,
                                      __bf16* __restrict__ xT) {
  int bg = blockIdx.x;
  int b = bg >> 5, g = bg & 31;
  int c0 = g * CPG;
  const size_t xbase = ((size_t)b * C_DIM + c0) * HW_DIM;
  const int NEl = CPG * HW_DIM;  // 32768

  float s = 0.f, s2 = 0.f;
  for (int idx = threadIdx.x; idx < NEl; idx += blockDim.x) {
    float v = x[xbase + idx];
    s += v; s2 += v * v;
  }
  __shared__ float red[512];
  red[threadIdx.x] = s;
  red[256 + threadIdx.x] = s2;
  __syncthreads();
  for (int off = 128; off > 0; off >>= 1) {
    if ((int)threadIdx.x < off) {
      red[threadIdx.x]       += red[threadIdx.x + off];
      red[256 + threadIdx.x] += red[256 + threadIdx.x + off];
    }
    __syncthreads();
  }
  float mean = red[0] * (1.0f / NEl);
  float var  = red[256] * (1.0f / NEl) - mean * mean;
  float rinv = rsqrtf(var + EPS_GN);

  for (int idx = threadIdx.x; idx < NEl; idx += blockDim.x) {
    int cc = idx >> 12;
    int i  = idx & 4095;
    int c  = c0 + cc;
    float v = x[xbase + idx];
    float hn = (v - mean) * rinv * gamma[c] + beta[c];
    size_t t = ((size_t)b * HW_DIM + i) * C_DIM + c;
    hnT[t] = (__bf16)hn;
    xT[t]  = (__bf16)v;
  }
}

// ---------------- Weight casts (layout unchanged: [out][in] row-major) ----------
__global__ void cast_weights_kernel(const float* __restrict__ Wq, const float* __restrict__ Wk,
                                    const float* __restrict__ Wv, const float* __restrict__ Wo,
                                    __bf16* __restrict__ Wqb, __bf16* __restrict__ Wkb,
                                    __bf16* __restrict__ Wvb, __bf16* __restrict__ Wob) {
  int i = blockIdx.x * blockDim.x + threadIdx.x;
  if (i < C_DIM * C_DIM) {
    Wqb[i] = (__bf16)Wq[i];
    Wkb[i] = (__bf16)Wk[i];
    Wvb[i] = (__bf16)Wv[i];
    Wob[i] = (__bf16)Wo[i];
  }
}

// ---------------- Generic WMMA GEMM: D = A(MxK rm) * Bc([N][K] rm) + bias -------
__global__ void gemm_bf16_kernel(const __bf16* __restrict__ A, const __bf16* __restrict__ Bc,
                                 const float* __restrict__ bias, __bf16* __restrict__ D,
                                 int lda, int ldb, int tiles_n64, int bias_on_m, int vmode) {
  int lane = threadIdx.x & 31;
  int w = blockIdx.x * (blockDim.x >> 5) + (threadIdx.x >> 5);
  int tm = w / tiles_n64;
  int tn = w - tm * tiles_n64;
  size_t m0 = (size_t)tm * 16;
  size_t n0 = (size_t)tn * 64;

  v8f acc[4];
#pragma unroll
  for (int t = 0; t < 4; ++t) acc[t] = zero8();

  for (int k0 = 0; k0 < C_DIM; k0 += 32) {
    v16bf a = load_a_frag(A, m0, lda, k0, lane);
#pragma unroll
    for (int t = 0; t < 4; ++t) {
      v16bf bb = load_b_frag(Bc, n0 + 16 * t, ldb, k0, lane);
      acc[t] = wmma_bf16(a, bb, acc[t]);
    }
  }

  int half = lane >> 4, nn = lane & 15;
#pragma unroll
  for (int t = 0; t < 4; ++t) {
#pragma unroll
    for (int r = 0; r < 8; ++r) {
      size_t m = m0 + (size_t)(r + 8 * half);
      size_t n = n0 + (size_t)(16 * t + nn);
      float val = acc[t][r] + (bias_on_m ? bias[m] : bias[n]);
      size_t addr;
      if (vmode) addr = ((n >> 12) << 20) + m * HW_DIM + (n & 4095);
      else       addr = m * C_DIM + n;
      D[addr] = (__bf16)val;
    }
  }
}

// ---------------- Flash attention with async double-buffered K/V staging --------
// All 8 waves of a block share one batch and stream identical K/V chunks, so the
// block stages each 32x256 K chunk + 256x32 V chunk into LDS once via
// global_load_async_to_lds_b128 (ping-pong buffers: async DMA of chunk j+1
// overlaps WMMA on chunk j), cutting L2 traffic 8x vs. per-wave loads.
__global__ void attn_flash_kernel(const __bf16* __restrict__ qT, const __bf16* __restrict__ kT,
                                  const __bf16* __restrict__ vmat, float* __restrict__ hT) {
  const int LDP = 36;
  __shared__ __bf16 kbuf[2][32 * 256];   // kT rows j0..j0+31 (contiguous 16KB)
  __shared__ __bf16 vbuf[2][256 * 32];   // v[c][j0..j0+31], 64B per channel row
  __shared__ __bf16 pbuf[8][16 * LDP];

  int lane = threadIdx.x & 31;
  int wid  = threadIdx.x >> 5;
  int tid  = threadIdx.x;
  int w  = blockIdx.x * 8 + wid;
  int b  = w >> 8;
  int i0 = (w & 255) * 16;
  __bf16* pb = &pbuf[wid][0];

  const __bf16* qbase = qT + ((size_t)b * HW_DIM + i0) * C_DIM;
  const __bf16* kbase = kT + (size_t)b * HW_DIM * C_DIM;
  const __bf16* vbase = vmat + (size_t)b * (size_t)C_DIM * HW_DIM;

  // stage one 32-key chunk of K and V into LDS buffer `buf` (whole block)
  auto stage = [&](int buf, int j0) {
    const char* ksrc = (const char*)(kbase + (size_t)j0 * C_DIM);
    unsigned klds = (unsigned)(size_t)(&kbuf[buf][0]);
#pragma unroll
    for (int q = 0; q < 4; ++q) {
      int idx = tid + q * 256;                 // 1024 x 16B = 16KB contiguous
      async_load_b128(klds + idx * 16, ksrc + idx * 16);
    }
    unsigned vlds = (unsigned)(size_t)(&vbuf[buf][0]);
#pragma unroll
    for (int q = 0; q < 4; ++q) {
      int idx = tid + q * 256;                 // idx = c*4 + col16
      int c = idx >> 2, col = idx & 3;
      const char* vsrc = (const char*)(vbase + (size_t)c * HW_DIM + j0) + col * 16;
      async_load_b128(vlds + idx * 16, vsrc);
    }
  };

  // q block resident as 8 A-fragments (read once)
  v16bf qa[8];
#pragma unroll
  for (int kk = 0; kk < 8; ++kk) qa[kk] = load_a_frag(qbase, 0, C_DIM, kk * 32, lane);

  v8f o[16];
#pragma unroll
  for (int t = 0; t < 16; ++t) o[t] = zero8();

  float mrow[8], lrow[8];
#pragma unroll
  for (int r = 0; r < 8; ++r) { mrow[r] = -3.0e38f; lrow[r] = 0.0f; }

  int half = lane >> 4, nn = lane & 15;
  const float SCALE = -0.5f * (float)C_DIM;  // reference multiplies by c * -0.5

  stage(0, 0);
  wait_async0();
  __syncthreads();

  for (int jc = 0; jc < HW_DIM / 32; ++jc) {
    int j0  = jc * 32;
    int cur = jc & 1;
    if (jc + 1 < HW_DIM / 32) stage(cur ^ 1, j0 + 32);  // overlap DMA with compute

    // ---- S tiles: 16 queries x 32 keys, K=256, B-frags from LDS ----
    const __bf16* kc = &kbuf[cur][0];
    v8f s0 = zero8(), s1 = zero8();
#pragma unroll
    for (int kk = 0; kk < 8; ++kk) {
      v16bf b0 = load_b_frag(kc, 0,  C_DIM, kk * 32, lane);
      v16bf b1 = load_b_frag(kc, 16, C_DIM, kk * 32, lane);
      s0 = wmma_bf16(qa[kk], b0, s0);
      s1 = wmma_bf16(qa[kk], b1, s1);
    }
    // ---- online softmax (row r+8*half across 16 lanes of each half) ----
#pragma unroll
    for (int r = 0; r < 8; ++r) {
      float a0 = s0[r] * SCALE, a1 = s1[r] * SCALE;
      float cm = fmaxf(a0, a1);
#pragma unroll
      for (int d = 1; d < 16; d <<= 1) cm = fmaxf(cm, __shfl_xor(cm, d, 16));
      float mn = fmaxf(mrow[r], cm);
      float alpha = __expf(mrow[r] - mn);
      mrow[r] = mn;
      float p0 = __expf(a0 - mn);
      float p1 = __expf(a1 - mn);
      s0[r] = p0; s1[r] = p1;
      float rs = p0 + p1;
#pragma unroll
      for (int d = 1; d < 16; d <<= 1) rs += __shfl_xor(rs, d, 16);
      lrow[r] = lrow[r] * alpha + rs;
#pragma unroll
      for (int t = 0; t < 16; ++t) o[t][r] *= alpha;
    }
    // ---- reshape P (D-layout) -> A-fragment via per-wave LDS staging ----
#pragma unroll
    for (int r = 0; r < 8; ++r) {
      int row = r + 8 * half;
      pb[row * LDP + nn]      = (__bf16)s0[r];
      pb[row * LDP + 16 + nn] = (__bf16)s1[r];
    }
    asm volatile("s_wait_dscnt 0x0" ::: "memory");  // DS in-order per wave
    v16bf pa = load_a_frag(pb, 0, LDP, 0, lane);
    asm volatile("" ::: "memory");
    // ---- O += P x V : Bc = vbuf[c][32], K = 32 keys ----
    const __bf16* vc = &vbuf[cur][0];
#pragma unroll
    for (int t = 0; t < 16; ++t) {
      v16bf bv = load_b_frag(vc, (size_t)t * 16, 32, 0, lane);
      o[t] = wmma_bf16(pa, bv, o[t]);
    }
    // my async writes for next chunk done; all waves done reading `cur`
    wait_async0();
    __syncthreads();
  }

  float inv[8];
#pragma unroll
  for (int r = 0; r < 8; ++r) inv[r] = 1.0f / lrow[r];
  float* hb = hT + ((size_t)b * HW_DIM + i0) * C_DIM;
#pragma unroll
  for (int t = 0; t < 16; ++t)
#pragma unroll
    for (int r = 0; r < 8; ++r)
      hb[(size_t)(r + 8 * half) * C_DIM + (size_t)(t * 16 + nn)] = o[t][r] * inv[r];
}

// ---------------- Final: out = h_ + Wo x + bo, transposed store to NCHW --------
__global__ void out_conv_add_kernel(const __bf16* __restrict__ xT, const __bf16* __restrict__ Wob,
                                    const float* __restrict__ bo, const float* __restrict__ hT,
                                    float* __restrict__ out) {
  int lane = threadIdx.x & 31;
  int w = blockIdx.x * 8 + (threadIdx.x >> 5);
  int tm = w >> 2;   // tiles_n64 == 4
  int tn = w & 3;
  size_t m0 = (size_t)tm * 16;
  size_t n0 = (size_t)tn * 64;

  v8f acc[4];
#pragma unroll
  for (int t = 0; t < 4; ++t) acc[t] = zero8();

  for (int k0 = 0; k0 < C_DIM; k0 += 32) {
    v16bf a = load_a_frag(xT, m0, C_DIM, k0, lane);
#pragma unroll
    for (int t = 0; t < 4; ++t) {
      v16bf bb = load_b_frag(Wob, n0 + 16 * t, C_DIM, k0, lane);
      acc[t] = wmma_bf16(a, bb, acc[t]);
    }
  }

  int half = lane >> 4, nn = lane & 15;
#pragma unroll
  for (int t = 0; t < 4; ++t) {
#pragma unroll
    for (int r = 0; r < 8; ++r) {
      size_t m = m0 + (size_t)(r + 8 * half);   // b*HW + i
      size_t n = n0 + (size_t)(16 * t + nn);    // channel c
      float val = acc[t][r] + bo[n] + hT[m * C_DIM + n];
      size_t bidx = m >> 12;
      size_t i    = m & 4095;
      out[(bidx << 20) + n * HW_DIM + i] = val;
    }
  }
}

extern "C" void kernel_launch(void* const* d_in, const int* in_sizes, int n_in,
                              void* d_out, int out_size, void* d_ws, size_t ws_size,
                              hipStream_t stream) {
  (void)in_sizes; (void)n_in; (void)out_size; (void)ws_size;
  const float* x     = (const float*)d_in[0];
  const float* gamma = (const float*)d_in[1];
  const float* beta  = (const float*)d_in[2];
  const float* Wq    = (const float*)d_in[3];
  const float* bq    = (const float*)d_in[4];
  const float* Wk    = (const float*)d_in[5];
  const float* bk    = (const float*)d_in[6];
  const float* Wv    = (const float*)d_in[7];
  const float* bv    = (const float*)d_in[8];
  const float* Wo    = (const float*)d_in[9];
  const float* bo    = (const float*)d_in[10];
  float* out = (float*)d_out;

  const size_t BF = (size_t)B_DIM * HW_DIM * C_DIM;   // 8M elements
  char* ws = (char*)d_ws;
  __bf16* hnT = (__bf16*)ws; ws += BF * 2;
  __bf16* xT  = (__bf16*)ws; ws += BF * 2;
  __bf16* qT  = (__bf16*)ws; ws += BF * 2;
  __bf16* kT  = (__bf16*)ws; ws += BF * 2;
  __bf16* vv  = (__bf16*)ws; ws += BF * 2;
  __bf16* Wqb = (__bf16*)ws; ws += (size_t)C_DIM * C_DIM * 2;
  __bf16* Wkb = (__bf16*)ws; ws += (size_t)C_DIM * C_DIM * 2;
  __bf16* Wvb = (__bf16*)ws; ws += (size_t)C_DIM * C_DIM * 2;
  __bf16* Wob = (__bf16*)ws; ws += (size_t)C_DIM * C_DIM * 2;
  float*  hT  = (float*)ws;  ws += BF * 4;

  cast_weights_kernel<<<256, 256, 0, stream>>>(Wq, Wk, Wv, Wo, Wqb, Wkb, Wvb, Wob);
  groupnorm_cast_kernel<<<B_DIM * NGROUPS, 256, 0, stream>>>(x, gamma, beta, hnT, xT);

  // qT[B*HW][C] = hnT x Wq : 2048 x 4 tiles = 8192 waves
  gemm_bf16_kernel<<<1024, 256, 0, stream>>>(hnT, Wqb, bq, qT, C_DIM, C_DIM, 4, 0, 0);
  // kT[B*HW][C] = hnT x Wk
  gemm_bf16_kernel<<<1024, 256, 0, stream>>>(hnT, Wkb, bk, kT, C_DIM, C_DIM, 4, 0, 0);
  // v[B][C][HW] = Wv x hnT : M=256 (16 tiles) x N=32768 (512 tiles of 64)
  gemm_bf16_kernel<<<1024, 256, 0, stream>>>(Wvb, hnT, bv, vv, C_DIM, C_DIM, 512, 1, 1);

  // flash attention: 8 batches x 256 query blocks = 2048 waves
  attn_flash_kernel<<<256, 256, 0, stream>>>(qT, kT, vv, hT);

  // out = h_ + conv1x1(x, Wo) + bo (transposed store to NCHW)
  out_conv_add_kernel<<<1024, 256, 0, stream>>>(xT, Wob, bo, hT, out);
}
